// OpenPanguEmbeddedAttention_29566554865742
// MI455X (gfx1250) — matrix operand order repaired and checked by reading, not compiled, not run on hardware
//
#include <hip/hip_runtime.h>
#include <hip/hip_bf16.h>

typedef __bf16 bf16;
typedef __attribute__((ext_vector_type(16))) __bf16 v16bf;
typedef __attribute__((ext_vector_type(8)))  __bf16 v8bf;
typedef __attribute__((ext_vector_type(8)))  float  v8f;
typedef __attribute__((ext_vector_type(4)))  int    v4i;

#define H_   32
#define KV_  8
#define D_   128
#define HID_ 4096
#define T_   2048
#define NQKV ((H_ + 2 * KV_) * D_)   // 6144

// ---------------------------------------------------------------------------
// Async global->LDS (ASYNCcnt path), guarded; fallback = sync copy via VGPR.
// Probe-confirmed prototype: (v4i as1*, v4i as3*, imm offset, imm cpol).
// ---------------------------------------------------------------------------
#if __has_builtin(__builtin_amdgcn_global_load_async_to_lds_b128) && \
    __has_builtin(__builtin_amdgcn_s_wait_asynccnt)
#define ASYNC_OK 1
#endif

#ifdef ASYNC_OK
__device__ __forceinline__ void async_copy16(const bf16* g, bf16* l) {
  __builtin_amdgcn_global_load_async_to_lds_b128(
      (__attribute__((address_space(1))) v4i*)(v4i*)(void*)g,
      (__attribute__((address_space(3))) v4i*)(v4i*)(void*)l, 0, 0);
}
#define WAIT_ASYNC(n) __builtin_amdgcn_s_wait_asynccnt(n)
#else
__device__ __forceinline__ void async_copy16(const bf16* g, bf16* l) {
  *(v8bf*)l = *(const v8bf*)g;
}
#define WAIT_ASYNC(n)
#endif

// ---------------------------------------------------------------------------
// WMMA fragment helpers (CDNA5 wave32 VGPR layouts, 16-bit operands)
// ---------------------------------------------------------------------------
__device__ __forceinline__ v16bf make16(v8bf lo, v8bf hi) {
  v16bf r;
#pragma unroll
  for (int i = 0; i < 8; ++i) { r[i] = lo[i]; r[i + 8] = hi[i]; }
  return r;
}

// A-matrix 16x32 (MxK), row-major source, leading dim ld (elements).
// Lane l: row m = l&15, kh = l>>4; 16B chunks at [k0+kh*8] and [k0+16+kh*8].
__device__ __forceinline__ v16bf frag_a(const bf16* p, int ld, int k0, int lane) {
  const int m = lane & 15, kh = lane >> 4;
  const bf16* q = p + (size_t)m * ld + k0 + kh * 8;
  return make16(*(const v8bf*)q, *(const v8bf*)(q + 16));
}

// B-matrix 32x16 (KxN) with B[k][n] = W[n][k], W row-major N x K (ld elements).
// Lane l: col n = n0 + (l&15); 16B chunks at [k0+kh*8] and [k0+16+kh*8], kh = l>>4.
__device__ __forceinline__ v16bf frag_b(const bf16* p, int ld, int n0, int k0, int lane) {
  const int n = n0 + (lane & 15), kh = lane >> 4;
  const bf16* q = p + (size_t)n * ld + k0 + kh * 8;
  return make16(*(const v8bf*)q, *(const v8bf*)(q + 16));
}

__device__ __forceinline__ v8f wmma_bf16(v16bf a, v16bf b, v8f c) {
  return __builtin_amdgcn_wmma_f32_16x16x32_bf16(false, a, false, b, (short)0, c,
                                                 false, false);
}

// ---------------------------------------------------------------------------
// fp32 -> bf16 cast
// ---------------------------------------------------------------------------
__global__ void cvt_f32_bf16(const float* __restrict__ in, bf16* __restrict__ out,
                             long n) {
  long i = ((long)blockIdx.x * blockDim.x + threadIdx.x) * 4;
  if (i + 3 < n) {
    float4 v = *(const float4*)(in + i);
    out[i]     = (bf16)v.x;
    out[i + 1] = (bf16)v.y;
    out[i + 2] = (bf16)v.z;
    out[i + 3] = (bf16)v.w;
  }
}

// ---------------------------------------------------------------------------
// C = A (MxK) * B^T (NxK), bf16 in, f32 accumulate.
// Block: 256 threads = 8 waves (2M x 4N); block tile 128x256, wave tile 64x64.
// Double-buffered async global->LDS staging, padded LDS rows (stride 40 el =
// 80B: 16B-aligned, bank shift 20 -> conflict-free 16-lane fragment reads).
// ---------------------------------------------------------------------------
template <bool STORE_BF16>
__global__ void __launch_bounds__(256)
gemm_bf16_nt(const bf16* __restrict__ A, const bf16* __restrict__ B,
             void* __restrict__ Cv, int M, int N, int K) {
  __shared__ __attribute__((aligned(16))) bf16 As[2][128 * 40];  // 2 x 10 KiB
  __shared__ __attribute__((aligned(16))) bf16 Bs[2][256 * 40];  // 2 x 20 KiB

  const int tid  = threadIdx.x;
  const int lane = tid & 31;
  const int w    = tid >> 5;
  const int mBlk = blockIdx.y * 128;
  const int nBlk = blockIdx.x * 256;
  const int mW   = (w >> 2) * 64;
  const int nW   = (w & 3) * 64;

  // stage one K-slice (32 wide): A 128x32, B 256x32; 6 async ops per thread
  auto stage = [&](int k0, int buf) {
#pragma unroll
    for (int c = tid; c < 512; c += 256) {           // A: 128 rows x 4 chunks
      const int row = c >> 2, ch = c & 3;
      async_copy16(A + (size_t)(mBlk + row) * K + k0 + ch * 8,
                   &As[buf][row * 40 + ch * 8]);
    }
#pragma unroll
    for (int c = tid; c < 1024; c += 256) {          // B: 256 rows x 4 chunks
      const int row = c >> 2, ch = c & 3;
      async_copy16(B + (size_t)(nBlk + row) * K + k0 + ch * 8,
                   &Bs[buf][row * 40 + ch * 8]);
    }
  };

  stage(0, 0);
  v8f acc[4][4] = {};
  const int nk = K / 32;
  for (int i = 0; i < nk; ++i) {
    if (i + 1 < nk) {
      stage((i + 1) * 32, (i + 1) & 1);   // prefetch next slice
      WAIT_ASYNC(6);                      // oldest 6 (slice i) complete
    } else {
      WAIT_ASYNC(0);
    }
    __syncthreads();

    const bf16* Ab = &As[i & 1][0];
    const bf16* Bb = &Bs[i & 1][0];
    v16bf af[4], bfr[4];
#pragma unroll
    for (int ii = 0; ii < 4; ++ii)
      af[ii] = frag_a(Ab + (size_t)(mW + ii * 16) * 40, 40, 0, lane);
#pragma unroll
    for (int j = 0; j < 4; ++j)
      bfr[j] = frag_b(Bb, 40, nW + j * 16, 0, lane);
#pragma unroll
    for (int ii = 0; ii < 4; ++ii)
#pragma unroll
      for (int j = 0; j < 4; ++j)
        acc[ii][j] = wmma_bf16(af[ii], bfr[j], acc[ii][j]);

    __syncthreads();   // all reads done before this buffer is refilled
  }

  const int kh = lane >> 4, nl = lane & 15;
#pragma unroll
  for (int i = 0; i < 4; ++i)
#pragma unroll
    for (int j = 0; j < 4; ++j)
#pragma unroll
      for (int r = 0; r < 8; ++r) {
        const int m = mBlk + mW + i * 16 + kh * 8 + r;
        const int n = nBlk + nW + j * 16 + nl;
        const float v = acc[i][j][r];
        if (STORE_BF16) ((bf16*)Cv)[(size_t)m * N + n] = (bf16)v;
        else            ((float*)Cv)[(size_t)m * N + n] = v;
      }
}

// ---------------------------------------------------------------------------
// NeoX RoPE, in place on the q/k slices of the bf16 qkv buffer (T x 6144).
// ---------------------------------------------------------------------------
__global__ void rope_kernel(const int* __restrict__ pos, bf16* __restrict__ qkv) {
  const int t = blockIdx.x;
  const float p = (float)pos[t];
  bf16* row = qkv + (size_t)t * NQKV;
  const float kLog = 0.14391156831212787f;          // ln(10000)/64
  for (int idx = threadIdx.x; idx < (H_ + KV_) * 64; idx += blockDim.x) {
    const int head = idx >> 6, j = idx & 63;
    const int base = (head < H_) ? head * D_ : H_ * D_ + (head - H_) * D_;
    const float inv = __expf(-(float)j * kLog);     // 10000^(-j/64)
    float s, c;
    __sincosf(p * inv, &s, &c);
    const float x1 = (float)row[base + j];
    const float x2 = (float)row[base + 64 + j];
    row[base + j]      = (bf16)(x1 * c - x2 * s);
    row[base + 64 + j] = (bf16)(x2 * c + x1 * s);
  }
}

// ---------------------------------------------------------------------------
// Causal flash attention with GQA (kv head = h/4), scale = 1/sqrt(128).
// Grid: (H, T/128); block 256 = 8 waves; wave owns 16 q rows.
// LDS rows padded for conflict-free fragment reads; K tile staged via async.
// ---------------------------------------------------------------------------
__global__ void __launch_bounds__(256)
attn_kernel(const bf16* __restrict__ qkv, bf16* __restrict__ out) {
  __shared__ __attribute__((aligned(16))) bf16 Kl[64 * 136];   // [s][d], 17 KiB
  __shared__ __attribute__((aligned(16))) bf16 Vt[128 * 72];   // [d][s], 18 KiB
  __shared__ __attribute__((aligned(16))) bf16 Pl[8][16 * 72]; // [m][s], 18 KiB

  const int h    = blockIdx.x;
  const int qt   = blockIdx.y;
  const int kvh  = h >> 2;
  const int lane = threadIdx.x & 31;
  const int w    = threadIdx.x >> 5;
  const int qrow = qt * 128 + w * 16;
  const int kh = lane >> 4, nl = lane & 15;

  // Q fragments for this wave's 16 rows, D=128 -> 4 K-steps of 32
  v16bf qf[4];
#pragma unroll
  for (int kk = 0; kk < 4; ++kk)
    qf[kk] = frag_a(qkv + (size_t)qrow * NQKV + h * D_, NQKV, kk * 32, lane);

  v8f   o[8] = {};
  float mx[8], sm[8];
#pragma unroll
  for (int r = 0; r < 8; ++r) { mx[r] = -1e30f; sm[r] = 0.f; }

  const float scale = 0.08838834764831845f;   // 1/sqrt(128)
  const int smax = qt * 128 + 127;

  for (int s0 = 0; s0 <= smax; s0 += 64) {
    __syncthreads();
    // cooperative K (async) / V^T (transposed) tile load: 64 keys x 128 dims
    for (int idx = threadIdx.x; idx < 64 * 16; idx += 256) {
      const int row = idx >> 4, ch = idx & 15;
      const bf16* src =
          qkv + (size_t)(s0 + row) * NQKV + H_ * D_ + kvh * D_ + ch * 8;
      async_copy16(src, &Kl[row * 136 + ch * 8]);
      v8bf vv = *(const v8bf*)(src + KV_ * D_);   // matching V chunk
#pragma unroll
      for (int e = 0; e < 8; ++e) Vt[(ch * 8 + e) * 72 + row] = vv[e];
    }
    WAIT_ASYNC(0);
    __syncthreads();

    // S = Q * K^T  (16 x 64), 4 n-subtiles x 4 K-steps
    v8f sacc[4] = {};
#pragma unroll
    for (int kk = 0; kk < 4; ++kk)
#pragma unroll
      for (int j = 0; j < 4; ++j) {
        v16bf kb = frag_b(Kl, 136, j * 16, kk * 32, lane);
        sacc[j] = wmma_bf16(qf[kk], kb, sacc[j]);
      }

    // scale + causal mask
#pragma unroll
    for (int j = 0; j < 4; ++j)
#pragma unroll
      for (int r = 0; r < 8; ++r) {
        const float v = sacc[j][r] * scale;
        const int qpos = qrow + kh * 8 + r;
        const int kpos = s0 + j * 16 + nl;
        sacc[j][r] = (kpos <= qpos) ? v : -1e30f;
      }

    // online softmax: row reductions across the 16-lane halves
#pragma unroll
    for (int r = 0; r < 8; ++r) {
      float rm = fmaxf(fmaxf(sacc[0][r], sacc[1][r]),
                       fmaxf(sacc[2][r], sacc[3][r]));
#pragma unroll
      for (int m_ = 1; m_ < 16; m_ <<= 1) rm = fmaxf(rm, __shfl_xor(rm, m_, 32));
      const float nm = fmaxf(mx[r], rm);
      const float alpha = __expf(mx[r] - nm);
      float rs = 0.f;
#pragma unroll
      for (int j = 0; j < 4; ++j) {
        const float e = __expf(sacc[j][r] - nm);
        sacc[j][r] = e;
        rs += e;
      }
#pragma unroll
      for (int m_ = 1; m_ < 16; m_ <<= 1) rs += __shfl_xor(rs, m_, 32);
      mx[r] = nm;
      sm[r] = sm[r] * alpha + rs;
#pragma unroll
      for (int dt = 0; dt < 8; ++dt) o[dt][r] *= alpha;
    }

    // probs -> LDS (per-wave, no barrier needed)
#pragma unroll
    for (int j = 0; j < 4; ++j)
#pragma unroll
      for (int r = 0; r < 8; ++r)
        Pl[w][(kh * 8 + r) * 72 + j * 16 + nl] = (bf16)sacc[j][r];

    // O += P * V  (2 K-steps over keys x 8 d-subtiles)
#pragma unroll
    for (int k2 = 0; k2 < 2; ++k2) {
      v16bf pa = frag_a(&Pl[w][0], 72, k2 * 32, lane);
#pragma unroll
      for (int dt = 0; dt < 8; ++dt) {
        v16bf vb = frag_b(Vt, 72, dt * 16, k2 * 32, lane);
        o[dt] = wmma_bf16(pa, vb, o[dt]);
      }
    }
  }

  // normalize and store (T x H*D, bf16)
#pragma unroll
  for (int dt = 0; dt < 8; ++dt)
#pragma unroll
    for (int r = 0; r < 8; ++r) {
      const float v = o[dt][r] / sm[r];
      const int t = qrow + kh * 8 + r;
      out[(size_t)t * (H_ * D_) + h * D_ + dt * 16 + nl] = (bf16)v;
    }
}

// ---------------------------------------------------------------------------
// Launch pipeline
// ---------------------------------------------------------------------------
extern "C" void kernel_launch(void* const* d_in, const int* in_sizes, int n_in,
                              void* d_out, int out_size, void* d_ws, size_t ws_size,
                              hipStream_t stream) {
  (void)in_sizes; (void)n_in; (void)out_size; (void)ws_size;
  const int*   positions = (const int*)d_in[0];
  const float* hidden    = (const float*)d_in[1];
  const float* wqkv      = (const float*)d_in[2];
  const float* wo        = (const float*)d_in[3];
  float* out = (float*)d_out;

  char* ws = (char*)d_ws;
  bf16* hid_bf  = (bf16*)(ws);                               // 16 MiB
  bf16* wqkv_bf = (bf16*)(ws + (size_t)16 * 1024 * 1024);    // 48 MiB
  bf16* wo_bf   = (bf16*)(ws + (size_t)64 * 1024 * 1024);    // 32 MiB
  bf16* qkv_bf  = (bf16*)(ws + (size_t)96 * 1024 * 1024);    // 24 MiB
  bf16* attn_bf = hid_bf;  // hidden_bf16 dead after GEMM1 -> reuse

  long n;
  n = (long)T_ * HID_;
  cvt_f32_bf16<<<(int)(n / 4 / 256), 256, 0, stream>>>(hidden, hid_bf, n);
  n = (long)NQKV * HID_;
  cvt_f32_bf16<<<(int)(n / 4 / 256), 256, 0, stream>>>(wqkv, wqkv_bf, n);
  n = (long)HID_ * (H_ * D_);
  cvt_f32_bf16<<<(int)(n / 4 / 256), 256, 0, stream>>>(wo, wo_bf, n);

  // qkv = hidden @ wqkv^T   (2048 x 6144, K = 4096)
  gemm_bf16_nt<true><<<dim3(NQKV / 256, T_ / 128), 256, 0, stream>>>(
      hid_bf, wqkv_bf, (void*)qkv_bf, T_, NQKV, HID_);

  rope_kernel<<<T_, 256, 0, stream>>>(positions, qkv_bf);

  attn_kernel<<<dim3(H_, T_ / 128), 256, 0, stream>>>(qkv_bf, attn_bf);

  // out = attn @ wo^T  (2048 x 4096, K = 4096), f32 output
  gemm_bf16_nt<false><<<dim3(HID_ / 256, T_ / 128), 256, 0, stream>>>(
      attn_bf, wo_bf, (void*)out, T_, HID_, H_ * D_);
}
